// CausalSelfAttention_84920093376827
// MI455X (gfx1250) — matrix operand order repaired and checked by reading, not compiled, not run on hardware
//
#include <hip/hip_runtime.h>

// ---------------- problem constants ----------------
#define B_   2
#define T_   2048
#define C_   768
#define H_   12
#define HD_  64

typedef __attribute__((ext_vector_type(16))) _Float16 v16h;
typedef __attribute__((ext_vector_type(8)))  float    v8f;

union FragU { v16h v; unsigned u[8]; _Float16 h[16]; };

// ------- WMMA fragment loaders (wave32, 16x16x32 f16) -------
// A fragment: rows m = lane%16; interleaved K pattern per ISA table:
//   elem pair i -> K = (i>=4?16:0) + half*8 + 2*(i&3) (+0/1)
__device__ __forceinline__ v16h load_a_f32(const float* row, int half) {
  FragU f;
#pragma unroll
  for (int i = 0; i < 8; ++i) {
    const int off = ((i >= 4) ? 16 : 0) + half * 8 + (i & 3) * 2;
    float2 t = *(const float2*)(row + off);
    f.h[2*i]   = (_Float16)t.x;
    f.h[2*i+1] = (_Float16)t.y;
  }
  return f.v;
}
__device__ __forceinline__ v16h load_a_f16(const _Float16* row, int half) {
  const unsigned* p = (const unsigned*)row;
  FragU f;
#pragma unroll
  for (int i = 0; i < 8; ++i)
    f.u[i] = p[((i >= 4) ? 8 : 0) + half * 4 + (i & 3)];
  return f.v;
}
// B fragment: cols n = lane%16; same interleaved K pattern as A
__device__ __forceinline__ v16h load_b_f32(const float* row, int half) {
  FragU f;
#pragma unroll
  for (int i = 0; i < 8; ++i) {
    float2 t = *(const float2*)(row + ((i >= 4) ? 16 : 0) + half * 8 + (i & 3) * 2);
    f.h[2*i]   = (_Float16)t.x;
    f.h[2*i+1] = (_Float16)t.y;
  }
  return f.v;
}
__device__ __forceinline__ v16h load_b_f16(const _Float16* row, int half) {
  const unsigned* p = (const unsigned*)row;
  FragU f;
#pragma unroll
  for (int i = 0; i < 8; ++i) f.u[i] = p[((i >= 4) ? 8 : 0) + half * 4 + (i & 3)];
  return f.v;
}

// =========== kernel 1: fused QKV projection (x @ W^T + b), 32x32/wave ===========
__global__ __launch_bounds__(128) void qkv_wmma_kernel(
    const float* __restrict__ x,
    const float* __restrict__ Wq, const float* __restrict__ bq,
    const float* __restrict__ Wk, const float* __restrict__ bk,
    const float* __restrict__ Wv, const float* __restrict__ bv,
    _Float16* __restrict__ Qo, _Float16* __restrict__ Ko, _Float16* __restrict__ Vo)
{
  const int lane = threadIdx.x & 31;
  const int half = lane >> 4;
  const int n    = lane & 15;
  const int gw   = blockIdx.x * 4 + (threadIdx.x >> 5);
  const int nRT = (B_ * T_) / 32, nCT = C_ / 32;     // 128, 24
  const int mat = gw / (nRT * nCT);
  const int rem = gw % (nRT * nCT);
  const int rt = rem / nCT, ct = rem % nCT;
  const int row0 = rt * 32, j0 = ct * 32;

  const float* W  = (mat == 0) ? Wq : ((mat == 1) ? Wk : Wv);
  const float* bs = (mat == 0) ? bq : ((mat == 1) ? bk : bv);

  // bias folded into accumulators: C-layout lane holds column j0 + y*16 + n
  v8f acc[2][2];
#pragma unroll
  for (int y = 0; y < 2; ++y) {
    const float bval = bs[j0 + y * 16 + n];
#pragma unroll
    for (int xq = 0; xq < 2; ++xq)
#pragma unroll
      for (int r = 0; r < 8; ++r) acc[xq][y][r] = bval;
  }

  const float* arow[2] = { x + (size_t)(row0 + n) * C_,
                           x + (size_t)(row0 + 16 + n) * C_ };
  const float* brow[2] = { W + (size_t)(j0 + n) * C_,
                           W + (size_t)(j0 + 16 + n) * C_ };
#pragma unroll 2
  for (int k0 = 0; k0 < C_; k0 += 32) {
    __builtin_prefetch(arow[0] + k0 + 64, 0, 3);   // global_prefetch_b8
    __builtin_prefetch(brow[0] + k0 + 64, 0, 3);
    v16h a0 = load_a_f32(arow[0] + k0, half);
    v16h a1 = load_a_f32(arow[1] + k0, half);
    v16h b0 = load_b_f32(brow[0] + k0, half);
    v16h b1 = load_b_f32(brow[1] + k0, half);
    acc[0][0] = __builtin_amdgcn_wmma_f32_16x16x32_f16(false, a0, false, b0, (short)0, acc[0][0], false, false);
    acc[0][1] = __builtin_amdgcn_wmma_f32_16x16x32_f16(false, a0, false, b1, (short)0, acc[0][1], false, false);
    acc[1][0] = __builtin_amdgcn_wmma_f32_16x16x32_f16(false, a1, false, b0, (short)0, acc[1][0], false, false);
    acc[1][1] = __builtin_amdgcn_wmma_f32_16x16x32_f16(false, a1, false, b1, (short)0, acc[1][1], false, false);
  }

#pragma unroll
  for (int xq = 0; xq < 2; ++xq) {
#pragma unroll
    for (int r = 0; r < 8; ++r) {
      const int rg = row0 + xq * 16 + r + half * 8;
      const int bI = rg >> 11, t = rg & (T_ - 1);
#pragma unroll
      for (int y = 0; y < 2; ++y) {
        const int j = j0 + y * 16 + n, h = j >> 6, d = j & 63;
        const _Float16 val = (_Float16)acc[xq][y][r];
        if (mat == 2)       // V transposed [B,H,64,T] for contiguous PV B-frags
          Vo[(((size_t)bI * H_ + h) * HD_ + d) * T_ + t] = val;
        else if (mat == 1)
          Ko[(((size_t)bI * H_ + h) * T_ + t) * HD_ + d] = val;
        else
          Qo[(((size_t)bI * H_ + h) * T_ + t) * HD_ + d] = val;
      }
    }
  }
}

// =========== kernel 2: flash attention, async K/V staging into LDS ===========
__global__ __launch_bounds__(128) void flash_attn_wmma_kernel(
    const _Float16* __restrict__ Q, const _Float16* __restrict__ K,
    const _Float16* __restrict__ Vt, const float* __restrict__ mask,
    _Float16* __restrict__ O)
{
  // per-wave slices: P tile (16x32), K tile (32x64), V tile (64x32)
  __shared__ __align__(16) _Float16 smem[4][16*32 + 32*64 + 64*32];
  const int lane  = threadIdx.x & 31;
  const int half  = lane >> 4;
  const int n     = lane & 15;
  const int wslot = threadIdx.x >> 5;
  const int gw    = blockIdx.x * 4 + wslot;
  const int nQT = T_ / 16;                       // 128
  const int bh = gw / nQT, qt = gw % nQT;
  const int b = bh / H_, h = bh % H_;
  const int q0 = qt * 16;

  _Float16* pL = smem[wslot];
  _Float16* kL = pL + 16 * 32;
  _Float16* vL = kL + 32 * 64;
  const unsigned kLa = (unsigned)(uintptr_t)kL;   // LDS byte offset (flat addr low bits)
  const unsigned vLa = (unsigned)(uintptr_t)vL;

  const _Float16* Qh = Q  + (size_t)bh * T_ * HD_;
  const _Float16* Kh = K  + (size_t)bh * T_ * HD_;
  const _Float16* Vh = Vt + (size_t)bh * HD_ * T_;
  const float*  mrow = mask + (size_t)b * T_ * T_;   // [B,1,T,T], head-broadcast

  const _Float16* qrow = Qh + (size_t)(q0 + n) * HD_;
  const v16h qa0 = load_a_f16(qrow,      half);      // d = 0..31
  const v16h qa1 = load_a_f16(qrow + 32, half);      // d = 32..63

  v8f o0 = {}, o1 = {}, o2 = {}, o3 = {};
  float Mr[8], Lr[8];
#pragma unroll
  for (int r = 0; r < 8; ++r) { Mr[r] = -__builtin_inff(); Lr[r] = 0.f; }

  const int nsteps = (q0 + 16 + 31) / 32;            // causal: keys 0 .. q0+15
  for (int ks = 0; ks < nsteps; ++ks) {
    const int kb = ks * 32;

    // ---- async-stage K tile (contiguous 4KB) and V tile (64 rows x 64B) ----
    asm volatile("s_wait_dscnt 0" ::: "memory");     // WAR: prior ds reads of kL/vL done
    {
      const char* ksrc = (const char*)(Kh + (size_t)kb * HD_);
#pragma unroll
      for (int i = 0; i < 8; ++i) {
        const int off = i * 512 + lane * 16;
        asm volatile("global_load_async_to_lds_b128 %0, %1, off"
                     :: "v"(kLa + off), "v"(ksrc + off) : "memory");
      }
      const char* vsrc = (const char*)(Vh + kb);
#pragma unroll
      for (int i = 0; i < 8; ++i) {
        const int c = i * 32 + lane;                 // 16B chunk id, 4 per row
        const int row = c >> 2, sub = (c & 3) * 16;
        asm volatile("global_load_async_to_lds_b128 %0, %1, off"
                     :: "v"(vLa + row * 64 + sub),
                        "v"(vsrc + (size_t)row * (T_ * 2) + sub) : "memory");
      }
    }
    asm volatile("s_wait_asynccnt 0" ::: "memory");

    // ---- S = Q K^T from LDS-resident K tile ----
    const _Float16* k0p = kL + (size_t)n * HD_;
    const _Float16* k1p = kL + (size_t)(16 + n) * HD_;
    v8f s0 = {}, s1 = {};
    s0 = __builtin_amdgcn_wmma_f32_16x16x32_f16(false, qa0, false, load_b_f16(k0p,      half), (short)0, s0, false, false);
    s0 = __builtin_amdgcn_wmma_f32_16x16x32_f16(false, qa1, false, load_b_f16(k0p + 32, half), (short)0, s0, false, false);
    s1 = __builtin_amdgcn_wmma_f32_16x16x32_f16(false, qa0, false, load_b_f16(k1p,      half), (short)0, s1, false, false);
    s1 = __builtin_amdgcn_wmma_f32_16x16x32_f16(false, qa1, false, load_b_f16(k1p + 32, half), (short)0, s1, false, false);

    // ---- causal + attn_mask, online softmax (row stats across 16-lane groups) ----
#pragma unroll
    for (int r = 0; r < 8; ++r) {
      const int m  = r + half * 8;
      const int qi = q0 + m;
      const int kA = kb + n, kB = kb + 16 + n;
      const float mA = mrow[(size_t)qi * T_ + kA];
      const float mB = mrow[(size_t)qi * T_ + kB];
      const float sA = ((kA <= qi) && (mA > 0.f)) ? s0[r] * 0.125f : -__builtin_inff();
      const float sB = ((kB <= qi) && (mB > 0.f)) ? s1[r] * 0.125f : -__builtin_inff();

      float mx = fmaxf(sA, sB);
#pragma unroll
      for (int off = 1; off < 16; off <<= 1) mx = fmaxf(mx, __shfl_xor(mx, off, 32));
      const float newM = fmaxf(Mr[r], mx);
      const float corr = __expf(Mr[r] - newM);
      const float pA = __expf(sA - newM);
      const float pB = __expf(sB - newM);
      float ps = pA + pB;
#pragma unroll
      for (int off = 1; off < 16; off <<= 1) ps += __shfl_xor(ps, off, 32);
      Lr[r] = Lr[r] * corr + ps;
      Mr[r] = newM;
      o0[r] *= corr; o1[r] *= corr; o2[r] *= corr; o3[r] *= corr;

      pL[m * 32 + n]      = (_Float16)pA;
      pL[m * 32 + 16 + n] = (_Float16)pB;
    }
    asm volatile("s_wait_dscnt 0" ::: "memory");     // wave-local LDS RAW on P tile
    const v16h pa = load_a_f16(pL + n * 32, half);   // P as A-fragment (16x32)

    // ---- O += P * V from LDS-resident V tile (row stride 32 halves) ----
    o0 = __builtin_amdgcn_wmma_f32_16x16x32_f16(false, pa, false, load_b_f16(vL + ( 0 + n) * 32, half), (short)0, o0, false, false);
    o1 = __builtin_amdgcn_wmma_f32_16x16x32_f16(false, pa, false, load_b_f16(vL + (16 + n) * 32, half), (short)0, o1, false, false);
    o2 = __builtin_amdgcn_wmma_f32_16x16x32_f16(false, pa, false, load_b_f16(vL + (32 + n) * 32, half), (short)0, o2, false, false);
    o3 = __builtin_amdgcn_wmma_f32_16x16x32_f16(false, pa, false, load_b_f16(vL + (48 + n) * 32, half), (short)0, o3, false, false);
  }

  // ---- normalize, write O in [B,T,C] f16 for the out-projection ----
  const int colBase = h * HD_;
#pragma unroll
  for (int r = 0; r < 8; ++r) {
    const int qi = q0 + r + half * 8;
    const float inv = 1.f / Lr[r];
    _Float16* orow = O + ((size_t)b * T_ + qi) * C_ + colBase;
    orow[ 0 + n] = (_Float16)(o0[r] * inv);
    orow[16 + n] = (_Float16)(o1[r] * inv);
    orow[32 + n] = (_Float16)(o2[r] * inv);
    orow[48 + n] = (_Float16)(o3[r] * inv);
  }
}

// =========== kernel 3: output projection (O @ Wp^T + bp) -> fp32, 32x32/wave ===========
__global__ __launch_bounds__(128) void proj_wmma_kernel(
    const _Float16* __restrict__ Oin, const float* __restrict__ Wp,
    const float* __restrict__ bp, float* __restrict__ Y)
{
  const int lane = threadIdx.x & 31;
  const int half = lane >> 4;
  const int n    = lane & 15;
  const int gw   = blockIdx.x * 4 + (threadIdx.x >> 5);
  const int nCT = C_ / 32;                       // 24
  const int rt = gw / nCT, ct = gw % nCT;
  const int row0 = rt * 32, j0 = ct * 32;

  v8f acc[2][2];
#pragma unroll
  for (int y = 0; y < 2; ++y) {
    const float bval = bp[j0 + y * 16 + n];
#pragma unroll
    for (int xq = 0; xq < 2; ++xq)
#pragma unroll
      for (int r = 0; r < 8; ++r) acc[xq][y][r] = bval;
  }

  const _Float16* arow[2] = { Oin + (size_t)(row0 + n) * C_,
                              Oin + (size_t)(row0 + 16 + n) * C_ };
  const float*    brow[2] = { Wp + (size_t)(j0 + n) * C_,
                              Wp + (size_t)(j0 + 16 + n) * C_ };
#pragma unroll 2
  for (int k0 = 0; k0 < C_; k0 += 32) {
    __builtin_prefetch(arow[0] + k0 + 64, 0, 3);
    __builtin_prefetch(brow[0] + k0 + 64, 0, 3);
    v16h a0 = load_a_f16(arow[0] + k0, half);
    v16h a1 = load_a_f16(arow[1] + k0, half);
    v16h b0 = load_b_f32(brow[0] + k0, half);
    v16h b1 = load_b_f32(brow[1] + k0, half);
    acc[0][0] = __builtin_amdgcn_wmma_f32_16x16x32_f16(false, a0, false, b0, (short)0, acc[0][0], false, false);
    acc[0][1] = __builtin_amdgcn_wmma_f32_16x16x32_f16(false, a0, false, b1, (short)0, acc[0][1], false, false);
    acc[1][0] = __builtin_amdgcn_wmma_f32_16x16x32_f16(false, a1, false, b0, (short)0, acc[1][0], false, false);
    acc[1][1] = __builtin_amdgcn_wmma_f32_16x16x32_f16(false, a1, false, b1, (short)0, acc[1][1], false, false);
  }

#pragma unroll
  for (int xq = 0; xq < 2; ++xq)
#pragma unroll
    for (int r = 0; r < 8; ++r) {
      const size_t rg = row0 + xq * 16 + r + half * 8;
#pragma unroll
      for (int y = 0; y < 2; ++y)
        Y[rg * C_ + j0 + y * 16 + n] = acc[xq][y][r];
    }
}

// =========== host-side launch ===========
extern "C" void kernel_launch(void* const* d_in, const int* in_sizes, int n_in,
                              void* d_out, int out_size, void* d_ws, size_t ws_size,
                              hipStream_t stream) {
  const float* x    = (const float*)d_in[0];
  const float* mask = (const float*)d_in[1];
  const float* Wq = (const float*)d_in[2]; const float* bq = (const float*)d_in[3];
  const float* Wk = (const float*)d_in[4]; const float* bk = (const float*)d_in[5];
  const float* Wv = (const float*)d_in[6]; const float* bv = (const float*)d_in[7];
  const float* Wp = (const float*)d_in[8]; const float* bp = (const float*)d_in[9];
  float* out = (float*)d_out;

  const size_t S = (size_t)B_ * H_ * T_ * HD_;   // 3,145,728 halves per tensor
  _Float16* ws = (_Float16*)d_ws;                // needs 4*S*2 = 24 MiB of workspace
  _Float16* Qb = ws;
  _Float16* Kb = ws + S;
  _Float16* Vb = ws + 2 * S;
  _Float16* Ob = ws + 3 * S;

  {
    const int waves = 3 * ((B_ * T_) / 32) * (C_ / 32);   // 9216
    qkv_wmma_kernel<<<dim3(waves / 4), dim3(128), 0, stream>>>(
        x, Wq, bq, Wk, bk, Wv, bv, Qb, Kb, Vb);
  }
  {
    const int waves = B_ * H_ * (T_ / 16);                // 3072
    flash_attn_wmma_kernel<<<dim3(waves / 4), dim3(128), 0, stream>>>(
        Qb, Kb, Vb, mask, Ob);
  }
  {
    const int waves = ((B_ * T_) / 32) * (C_ / 32);       // 3072
    proj_wmma_kernel<<<dim3(waves / 4), dim3(128), 0, stream>>>(
        Ob, Wp, bp, out);
  }
}
